// BigramLanguageModel_84980222918885
// MI455X (gfx1250) — matrix operand order repaired and checked by reading, not compiled, not run on hardware
//
#include <hip/hip_runtime.h>
#include <hip/hip_bf16.h>

typedef __bf16 bf16;
typedef __attribute__((ext_vector_type(16))) __bf16 v16bf;
typedef __attribute__((ext_vector_type(8)))  __bf16 v8bf;
typedef __attribute__((ext_vector_type(8)))  float  v8f;
typedef unsigned u32;
typedef __attribute__((ext_vector_type(4))) u32 u32x4;
typedef __attribute__((ext_vector_type(8))) u32 u32x8;

static constexpr int Bc  = 4;
static constexpr int Tc  = 1024;
static constexpr int Dc  = 1024;
static constexpr int Hc  = 4;
static constexpr int DHc = 256;     // Dc / Hc
static constexpr int FFc = 4096;
static constexpr int Vc  = 50257;
static constexpr int BTc = 4096;    // Bc * Tc

// Output modes for the macro-GEMM
static constexpr int OM_F32  = 0;   // f32 out, C[m*ldc+n]
static constexpr int OM_BF16 = 1;   // bf16 out, C[m*ldc+n]
static constexpr int OM_KT   = 2;   // bf16 out transposed per-batch: kT[H][B][DH][T]

// LDS byte-offset of a __shared__ pointer: generic LDS addresses carry the
// byte offset in the low 32 bits (aperture in [63:32]), so truncation works.
__device__ __forceinline__ unsigned lds_off(const void* p) {
    return (unsigned)(unsigned long long)p;
}

// Async direct-to-LDS 128-bit load (CDNA5 GLOBAL_LOAD_ASYNC_TO_LDS_B128,
// tracked by ASYNCcnt; no VGPR staging).
__device__ __forceinline__ void async_load_b128(unsigned lds_addr, const void* gaddr) {
    asm volatile("global_load_async_to_lds_b128 %0, %1, off"
                 :: "v"(lds_addr), "v"((unsigned long long)gaddr)
                 : "memory");
}

__device__ __forceinline__ void wait_async0() {
    asm volatile("s_wait_asynccnt 0" ::: "memory");
}

// ---------------------------------------------------------------------------
// TDM: DMA a 2D tile (tile_d1 rows x tile_d0 elements, 2-byte elements) from
// global memory into LDS, with hardware row padding in LDS and hardware
// zero-fill for reads beyond tensor_d0/tensor_d1 (handles the ragged LM-head
// edge and its unaligned row stride for free). Descriptor per ISA 08 §8.
// Issued once per wave; tracked by TENSORcnt.
// ---------------------------------------------------------------------------
__device__ __forceinline__ void tdm_load_2d_b16(
    unsigned lds_addr, const void* gaddr,
    u32 tensor_d0, u32 tensor_d1, u32 tile_d0, u32 tile_d1,
    unsigned long long stride0_elems, u32 pad_interval, u32 pad_amount)
{
    const unsigned long long ga = (unsigned long long)gaddr;
    u32x4 g0;
    g0[0] = 1u;                                            // count=1, user D#
    g0[1] = lds_addr;                                      // LDS byte address
    g0[2] = (u32)ga;                                       // global_addr[31:0]
    g0[3] = ((u32)(ga >> 32) & 0x01FFFFFFu) | 0x80000000u; // addr[56:32] | type=2
    u32x8 g1;
    g1[0] = (1u << 16)                 // data_size = 1 (2 bytes)
          | (1u << 20)                 // pad_enable
          | (pad_interval << 22)       // DWORDs between pads = 2<<interval
          | (pad_amount   << 25);      // pad DWORDs = amount+1
    g1[1] = (tensor_d0 & 0xFFFFu) << 16;                       // dim0[15:0]
    g1[2] = (tensor_d0 >> 16) | ((tensor_d1 & 0xFFFFu) << 16); // dim0[31:16], dim1[15:0]
    g1[3] = (tensor_d1 >> 16) | (tile_d0 << 16);               // dim1[31:16], tile0
    g1[4] = tile_d1;                                           // tile1 (tile2=0 -> 2D)
    g1[5] = (u32)stride0_elems;                                // dim0_stride[31:0]
    g1[6] = (u32)(stride0_elems >> 32) & 0xFFFFu;              // dim0_stride[47:32]
    g1[7] = 0u;
    asm volatile("tensor_load_to_lds %0, %1" :: "s"(g0), "s"(g1) : "memory");
}

// ---------------------------------------------------------------------------
// Generic bf16 WMMA GEMM: C = A[M,K] * B[K,N] (+bias) (optional ReLU)
// Block tile 128x128, 8 waves (wave32), wave tile 32x64 (2x4 16x16 frags),
// K-step 32, double-buffered LDS staging: A via per-lane async direct-to-LDS
// loads, B via one TDM tensor_load_to_lds per tile (wave 0).
// Batched via grid.z: off = (z/zdiv)*hi + (z%zdiv)*lo  (element strides).
// ---------------------------------------------------------------------------
template<int OUTMODE, bool RELU, bool NCHECK>
__global__ __launch_bounds__(256) void gemm_wmma(
    const bf16* __restrict__ A, long lda,
    const bf16* __restrict__ Bm, long ldb,
    void* __restrict__ Cv, long ldc,
    const float* __restrict__ bias,
    int M, int N, int K, int zdiv,
    long zAhi, long zAlo, long zBhi, long zBlo, long zChi, long zClo)
{
    constexpr int BM = 128, BN = 128, BK = 32;
    constexpr int SA = BK + 8;   // padded LDS row stride (bf16) for A tile
    constexpr int SB = BN + 8;   // padded LDS row stride (bf16) for B tile
    __shared__ bf16 sA[2][BM * SA];
    __shared__ bf16 sB[2][BK * SB];

    const int z  = blockIdx.z;
    const long zh = z / zdiv, zl = z % zdiv;
    A  += zh * zAhi + zl * zAlo;
    Bm += zh * zBhi + zl * zBlo;
    const long cbase = zh * zChi + zl * zClo;

    const int tid  = threadIdx.x;
    const int lane = tid & 31;
    const int wid  = tid >> 5;
    const int wm   = wid & 3;    // 4 waves down M (32 rows each)
    const int wn   = wid >> 2;   // 2 waves across N (64 cols each)

    const int m0 = blockIdx.y * BM;
    const int n0 = blockIdx.x * BN;

    v8f acc[2][4];
#pragma unroll
    for (int i = 0; i < 2; ++i)
#pragma unroll
        for (int j = 0; j < 4; ++j)
#pragma unroll
            for (int e = 0; e < 8; ++e) acc[i][j][e] = 0.f;

    const int nK = K >> 5;

    auto load_tiles = [&](int kt, int buf) {
        const int k0 = kt << 5;
        // A tile: 128x32 bf16 = 512 x 16B vectors, 2 per thread, async -> LDS
#pragma unroll
        for (int i = 0; i < 2; ++i) {
            const int vi = tid * 2 + i;
            const int r  = vi >> 2;
            const int c  = (vi & 3) << 3;
            async_load_b128(lds_off(&sA[buf][r * SA + c]),
                            A + (long)(m0 + r) * lda + (k0 + c));
        }
        // B tile: 32 rows x 128 cols bf16, one TDM descriptor issued by wave 0.
        // Rows: 128 bf16 = 64 DWORDs, then 4 DWORDs pad -> LDS stride SB=136.
        // tensor_d0 = N-n0 gives hardware zero-fill at the ragged N edge.
        if (wid == 0) {
            tdm_load_2d_b16(lds_off(&sB[buf][0]),
                            Bm + (long)k0 * ldb + n0,
                            (u32)(N - n0), (u32)(K - k0),
                            (u32)BN, (u32)BK,
                            (unsigned long long)ldb,
                            /*pad_interval=*/5u /*64 DW*/, /*pad_amount=*/3u /*4 DW*/);
        }
    };

    load_tiles(0, 0);
    wait_async0();
    if (wid == 0) __builtin_amdgcn_s_wait_tensorcnt(0);
    __syncthreads();

    for (int kt = 0; kt < nK; ++kt) {
        if (kt + 1 < nK) load_tiles(kt + 1, (kt + 1) & 1);
        if (kt + 2 < nK) {
            const int k2 = (kt + 2) << 5;
            __builtin_prefetch(A + (long)(m0 + (tid >> 1)) * lda + k2, 0, 1);
        }
        const int buf = kt & 1;

        // A fragments: lane holds row (lane&15); K-halves chosen by lane>>4.
        v16bf af[2], bfr[4];
        const int rowl = lane & 15;
        const int kb   = (lane >> 4) << 3;   // 0 or 8
#pragma unroll
        for (int fm = 0; fm < 2; ++fm) {
            const bf16* base = &sA[buf][(wm * 32 + fm * 16 + rowl) * SA + kb];
            v8bf lo = *(const v8bf*)(base);
            v8bf hi = *(const v8bf*)(base + 16);
#pragma unroll
            for (int e = 0; e < 8; ++e) { af[fm][e] = lo[e]; af[fm][e + 8] = hi[e]; }
        }
        // B fragments: lane holds column N=(lane&15); K-halves chosen by lane>>4.
#pragma unroll
        for (int fn = 0; fn < 4; ++fn) {
            const bf16* base = &sB[buf][kb * SB + wn * 64 + fn * 16 + rowl];
#pragma unroll
            for (int e = 0; e < 8; ++e) { bfr[fn][e] = base[e * SB]; bfr[fn][e + 8] = base[(16 + e) * SB]; }
        }
#pragma unroll
        for (int fm = 0; fm < 2; ++fm)
#pragma unroll
            for (int fn = 0; fn < 4; ++fn)
                acc[fm][fn] = __builtin_amdgcn_wmma_f32_16x16x32_bf16(
                    false, af[fm], false, bfr[fn], (short)0, acc[fm][fn], false, false);

        wait_async0();                                    // A tile landed
        if (wid == 0) __builtin_amdgcn_s_wait_tensorcnt(0); // B tile landed
        __syncthreads();
    }

    // Epilogue: C/D layout -> row = j + 8*(lane>>4), col = lane&15
#pragma unroll
    for (int fm = 0; fm < 2; ++fm) {
#pragma unroll
        for (int fn = 0; fn < 4; ++fn) {
#pragma unroll
            for (int j = 0; j < 8; ++j) {
                const int m = m0 + wm * 32 + fm * 16 + j + ((lane >> 4) << 3);
                const int n = n0 + wn * 64 + fn * 16 + (lane & 15);
                if (NCHECK && n >= N) continue;
                float vv = acc[fm][fn][j];
                if (bias) vv += bias[n];
                if (RELU) vv = fmaxf(vv, 0.f);
                if (OUTMODE == OM_F32) {
                    ((float*)Cv)[cbase + (long)m * ldc + n] = vv;
                } else if (OUTMODE == OM_BF16) {
                    ((bf16*)Cv)[cbase + (long)m * ldc + n] = (bf16)vv;
                } else { // OM_KT: dest[b][n][t], b=m/T, t=m%T, per-b stride N*T
                    ((bf16*)Cv)[cbase + (long)(m >> 10) * (((long)N) << 10)
                                      + (((long)n) << 10) + (m & (Tc - 1))] = (bf16)vv;
                }
            }
        }
    }
}

// ---------------------------------------------------------------------------
// h[bt,d] = tok_emb[x[bt],d] + pos_emb[t,d]   (bf16 out)
// ---------------------------------------------------------------------------
__global__ __launch_bounds__(256) void embed_kernel(
    const int* __restrict__ x, const float* __restrict__ tok,
    const float* __restrict__ pos, bf16* __restrict__ h)
{
    const long i  = (long)blockIdx.x * 256 + threadIdx.x;   // < BTc*Dc
    const int  d  = (int)(i & (Dc - 1));
    const int  bt = (int)(i >> 10);
    const int  t  = bt & (Tc - 1);
    const int  idx = x[bt];
    h[i] = (bf16)(tok[(long)idx * Dc + d] + pos[(long)t * Dc + d]);
}

__global__ __launch_bounds__(256) void f32_to_bf16(
    const float* __restrict__ in, bf16* __restrict__ out, long n)
{
    const long i = (long)blockIdx.x * 256 + threadIdx.x;
    if (i < n) out[i] = (bf16)in[i];
}

// ---------------------------------------------------------------------------
// Causal softmax over one row of S (length T), scaled; bf16 probs out.
// One block (256 thr) per row; fixed-order tree reductions (deterministic).
// ---------------------------------------------------------------------------
__global__ __launch_bounds__(256) void softmax_causal(
    const float* __restrict__ S, bf16* __restrict__ P, float scale)
{
    __shared__ float red[256];
    const int row = blockIdx.x;
    const int tq  = row & (Tc - 1);
    const float* s = S + (long)row * Tc;
    bf16* p = P + (long)row * Tc;
    const int tid = threadIdx.x;

    float mx = -__builtin_inff();
    for (int j = tid; j < Tc; j += 256)
        if (j <= tq) mx = fmaxf(mx, s[j]);
    red[tid] = mx; __syncthreads();
    for (int st = 128; st > 0; st >>= 1) {
        if (tid < st) red[tid] = fmaxf(red[tid], red[tid + st]);
        __syncthreads();
    }
    mx = red[0] * scale;
    __syncthreads();

    float sum = 0.f;
    float ev[Tc / 256];
    int idx = 0;
    for (int j = tid; j < Tc; j += 256, ++idx) {
        const float e = (j <= tq) ? __expf(s[j] * scale - mx) : 0.f;
        ev[idx] = e; sum += e;
    }
    red[tid] = sum; __syncthreads();
    for (int st = 128; st > 0; st >>= 1) {
        if (tid < st) red[tid] += red[tid + st];
        __syncthreads();
    }
    const float inv = 1.f / red[0];
    idx = 0;
    for (int j = tid; j < Tc; j += 256, ++idx) p[j] = (bf16)(ev[idx] * inv);
}

// ---------------------------------------------------------------------------
// Per-row NLL over logits [BTc, Vc]; one block per row; deterministic.
// ---------------------------------------------------------------------------
__global__ __launch_bounds__(256) void nll_rows(
    const float* __restrict__ logits, const int* __restrict__ y,
    float* __restrict__ nll)
{
    __shared__ float red[256];
    const int row = blockIdx.x;
    const float* l = logits + (long)row * Vc;
    const int tid = threadIdx.x;

    float mx = -__builtin_inff();
    for (int j = tid; j < Vc; j += 256) mx = fmaxf(mx, l[j]);
    red[tid] = mx; __syncthreads();
    for (int st = 128; st > 0; st >>= 1) {
        if (tid < st) red[tid] = fmaxf(red[tid], red[tid + st]);
        __syncthreads();
    }
    mx = red[0]; __syncthreads();

    float sum = 0.f;
    for (int j = tid; j < Vc; j += 256) sum += __expf(l[j] - mx);
    red[tid] = sum; __syncthreads();
    for (int st = 128; st > 0; st >>= 1) {
        if (tid < st) red[tid] += red[tid + st];
        __syncthreads();
    }
    if (tid == 0) {
        const float lse = mx + __logf(red[0]);
        nll[row] = lse - l[y[row]];
    }
}

__global__ __launch_bounds__(256) void loss_reduce(
    const float* __restrict__ nll, float* __restrict__ out)
{
    __shared__ float red[256];
    const int tid = threadIdx.x;
    float s = 0.f;
    for (int k = 0; k < BTc / 256; ++k) s += nll[tid + k * 256];
    red[tid] = s; __syncthreads();
    for (int st = 128; st > 0; st >>= 1) {
        if (tid < st) red[tid] += red[tid + st];
        __syncthreads();
    }
    if (tid == 0) out[0] = red[0] / (float)BTc;
}

// ---------------------------------------------------------------------------
extern "C" void kernel_launch(void* const* d_in, const int* in_sizes, int n_in,
                              void* d_out, int out_size, void* d_ws, size_t ws_size,
                              hipStream_t stream) {
    (void)in_sizes; (void)n_in; (void)out_size; (void)ws_size;
    const int*   x    = (const int*)  d_in[0];
    const int*   y    = (const int*)  d_in[1];
    const float* tok  = (const float*)d_in[2];
    const float* pos  = (const float*)d_in[3];
    const float* Wq   = (const float*)d_in[4];
    const float* Wk   = (const float*)d_in[5];
    const float* Wv   = (const float*)d_in[6];
    const float* W1   = (const float*)d_in[7];
    const float* b1   = (const float*)d_in[8];
    const float* W2   = (const float*)d_in[9];
    const float* b2   = (const float*)d_in[10];
    const float* Wlm  = (const float*)d_in[11];
    const float* blm  = (const float*)d_in[12];
    float* out = (float*)d_out;

    char* wsp = (char*)d_ws;
    auto alloc = [&](size_t bytes) -> void* {
        void* p = wsp; wsp += (bytes + 255) & ~(size_t)255; return p;
    };
    const size_t nWqkv = (size_t)Hc * Dc * DHc;
    bf16* Wqbf  = (bf16*)alloc(2 * nWqkv);
    bf16* Wkbf  = (bf16*)alloc(2 * nWqkv);
    bf16* Wvbf  = (bf16*)alloc(2 * nWqkv);
    bf16* W1bf  = (bf16*)alloc(2 * (size_t)Dc * FFc);
    bf16* W2bf  = (bf16*)alloc(2 * (size_t)FFc * Dc);
    bf16* Wlmbf = (bf16*)alloc(2 * (size_t)Dc * Vc);
    bf16* hbf   = (bf16*)alloc(2 * (size_t)BTc * Dc);
    bf16* qbf   = (bf16*)alloc(2 * (size_t)Hc * BTc * DHc);   // [H][BT][DH]
    bf16* kTbf  = (bf16*)alloc(2 * (size_t)Hc * BTc * DHc);   // [H][B][DH][T]
    bf16* vbf   = (bf16*)alloc(2 * (size_t)Hc * BTc * DHc);   // [H][BT][DH]
    float* Sf   = (float*)alloc(4 * (size_t)Bc * Hc * Tc * Tc); // [B][H][T][T]
    bf16* Pbf   = (bf16*)alloc(2 * (size_t)Bc * Hc * Tc * Tc);
    bf16* obf   = (bf16*)alloc(2 * (size_t)BTc * Dc);
    bf16* ffbf  = (bf16*)alloc(2 * (size_t)BTc * FFc);
    bf16* fbf   = (bf16*)alloc(2 * (size_t)BTc * Dc);
    float* nllb = (float*)alloc(4 * (size_t)BTc);

    auto cvt = [&](const float* in, bf16* o, long n) {
        f32_to_bf16<<<(unsigned)((n + 255) / 256), 256, 0, stream>>>(in, o, n);
    };
    cvt(Wq, Wqbf, (long)nWqkv);
    cvt(Wk, Wkbf, (long)nWqkv);
    cvt(Wv, Wvbf, (long)nWqkv);
    cvt(W1, W1bf, (long)Dc * FFc);
    cvt(W2, W2bf, (long)FFc * Dc);
    cvt(Wlm, Wlmbf, (long)Dc * Vc);

    embed_kernel<<<(BTc * Dc) / 256, 256, 0, stream>>>(x, tok, pos, hbf);

    // ---- QKV projections (batched over heads via grid.z) ----
    const long hdW = (long)Dc * DHc;     // head stride in Wq/Wk/Wv
    const long hdQ = (long)BTc * DHc;    // head stride in q/k/v
    gemm_wmma<OM_BF16, false, false><<<dim3(DHc/128, BTc/128, Hc), 256, 0, stream>>>(
        hbf, Dc, Wqbf, DHc, qbf, DHc, nullptr, BTc, DHc, Dc,
        1, 0, 0, hdW, 0, hdQ, 0);
    gemm_wmma<OM_KT, false, false><<<dim3(DHc/128, BTc/128, Hc), 256, 0, stream>>>(
        hbf, Dc, Wkbf, DHc, kTbf, DHc, nullptr, BTc, DHc, Dc,
        1, 0, 0, hdW, 0, hdQ, 0);
    gemm_wmma<OM_BF16, false, false><<<dim3(DHc/128, BTc/128, Hc), 256, 0, stream>>>(
        hbf, Dc, Wvbf, DHc, vbf, DHc, nullptr, BTc, DHc, Dc,
        1, 0, 0, hdW, 0, hdQ, 0);

    // ---- scores S[b,h] = q[h][b] * kT[h][b] : grid.z = b*H + h ----
    gemm_wmma<OM_F32, false, false><<<dim3(Tc/128, Tc/128, Bc*Hc), 256, 0, stream>>>(
        qbf, DHc, kTbf, Tc, Sf, Tc, nullptr, Tc, Tc, DHc, Hc,
        (long)Tc * DHc, (long)BTc * DHc,          // A: b-stride, h-stride
        (long)DHc * Tc, (long)Bc * DHc * Tc,      // B: b-stride, h-stride
        (long)Hc * Tc * Tc, (long)Tc * Tc);       // C: b-stride, h-stride

    softmax_causal<<<Bc * Hc * Tc, 256, 0, stream>>>(Sf, Pbf, 0.03125f); // D^-0.5

    // ---- O[b,h] = P[b,h] * v[h][b], stored concat-heads into obf[BT,D] ----
    gemm_wmma<OM_BF16, false, false><<<dim3(DHc/128, Tc/128, Bc*Hc), 256, 0, stream>>>(
        Pbf, Tc, vbf, DHc, obf, Dc, nullptr, Tc, DHc, Tc, Hc,
        (long)Hc * Tc * Tc, (long)Tc * Tc,
        (long)Tc * DHc, (long)BTc * DHc,
        (long)Tc * Dc, (long)DHc);

    // ---- FFN: relu(o@W1+b1)@W2+b2 ----
    gemm_wmma<OM_BF16, true, false><<<dim3(FFc/128, BTc/128, 1), 256, 0, stream>>>(
        obf, Dc, W1bf, FFc, ffbf, FFc, b1, BTc, FFc, Dc, 1, 0,0,0,0,0,0);
    gemm_wmma<OM_BF16, false, false><<<dim3(Dc/128, BTc/128, 1), 256, 0, stream>>>(
        ffbf, FFc, W2bf, Dc, fbf, Dc, b2, BTc, Dc, FFc, 1, 0,0,0,0,0,0);

    // ---- LM head: logits = f @ Wlm + blm  (f32 straight into d_out) ----
    // TDM zero-fills the ragged N edge; epilogue bounds-checks stores.
    gemm_wmma<OM_F32, false, true><<<dim3((Vc + 127) / 128, BTc/128, 1), 256, 0, stream>>>(
        fbf, Dc, Wlmbf, Vc, out, Vc, blm, BTc, Vc, Dc, 1, 0,0,0,0,0,0);

    // ---- loss = mean over rows of (logsumexp - logit[y]) ----
    nll_rows<<<BTc, 256, 0, stream>>>(out, y, nllb);
    loss_reduce<<<1, 256, 0, stream>>>(nllb, out + (long)BTc * Vc);
}
